// ResidualAttentionLayer_8074538517040
// MI455X (gfx1250) — compile-verified, matrix operand order checked by reading, not run on hardware
//
#include <hip/hip_runtime.h>
#include <hip/hip_bf16.h>

// ---------------- types / helpers ----------------
typedef __attribute__((ext_vector_type(16))) __bf16 v16bf;
typedef __attribute__((ext_vector_type(8)))  float  v8f;
typedef __attribute__((ext_vector_type(4)))  unsigned u32x4;
typedef __attribute__((ext_vector_type(8)))  int      i32x8;
typedef __attribute__((ext_vector_type(4)))  int      i32x4;

union BFrag { v16bf v; __bf16 e[16]; uint4 q[2]; };

static __device__ __forceinline__ __bf16 f2bf(float f) {
    union { float f; unsigned u; } cf; cf.f = f;
    unsigned r = cf.u + 0x7FFFu + ((cf.u >> 16) & 1u);   // round-to-nearest-even
    union { unsigned short s; __bf16 b; } cb; cb.s = (unsigned short)(r >> 16);
    return cb.b;
}

// problem constants
static constexpr int kB   = 2;
static constexpr int kT   = 2048;
static constexpr int kD   = 1024;
static constexpr int kH   = 16;
static constexpr int kDH  = 64;
static constexpr int kD3  = 3 * kD;
static constexpr int kDFF = 4 * kD;

// ================= weight convert + transpose =================
// in: f32 [Kd, Nd] row-major.  out: bf16 [Nd, Kd] row-major (out[n*Kd+k] = in[k*Nd+n])
__global__ __launch_bounds__(256)
void k_cvt_transpose(const float* __restrict__ in, __bf16* __restrict__ out,
                     int Kd, int Nd) {
    __shared__ float tile[32][33];
    int bx = blockIdx.x * 32;             // n base
    int by = blockIdx.y * 32;             // k base
    int tx = threadIdx.x & 31;
    int ty = threadIdx.x >> 5;            // 0..7
    #pragma unroll
    for (int r = 0; r < 32; r += 8)
        tile[ty + r][tx] = in[(size_t)(by + ty + r) * Nd + bx + tx];
    __syncthreads();
    #pragma unroll
    for (int r = 0; r < 32; r += 8)
        out[(size_t)(bx + ty + r) * Kd + by + tx] = f2bf(tile[tx][ty + r]);
}

// ================= V transpose: qkv -> vT[bh][n][k] =================
__global__ __launch_bounds__(256)
void k_transpose_v(const __bf16* __restrict__ qkv, __bf16* __restrict__ vT) {
    __shared__ __bf16 t[64][65];
    int bh = blockIdx.z; int b = bh >> 4, h = bh & 15;
    int k0 = blockIdx.x * 64;
    int tx = threadIdx.x & 63;
    int ty = threadIdx.x >> 6;            // 0..3
    size_t vbase = (size_t)b * kT * kD3 + 2 * kD + (size_t)h * kDH;
    #pragma unroll
    for (int r = 0; r < 64; r += 4)
        t[ty + r][tx] = qkv[vbase + (size_t)(k0 + ty + r) * kD3 + tx];
    __syncthreads();
    size_t obase = (size_t)bh * kDH * kT;
    #pragma unroll
    for (int r = 0; r < 64; r += 4)
        vT[obase + (size_t)(ty + r) * kT + k0 + tx] = t[tx][ty + r];
}

// ================= LayerNorm (row of D=1024) -> bf16 =================
__global__ __launch_bounds__(256)
void k_ln_bf16(const float* __restrict__ x, const float* __restrict__ g,
               const float* __restrict__ be, __bf16* __restrict__ out) {
    const int Dl = kD;
    int row = blockIdx.x;
    int tid = threadIdx.x;
    const float* xr = x + (size_t)row * Dl;
    __shared__ float red[256];

    float lx[4]; float s = 0.f;
    #pragma unroll
    for (int j = 0; j < 4; ++j) { lx[j] = xr[tid + 256 * j]; s += lx[j]; }
    red[tid] = s; __syncthreads();
    for (int off = 128; off > 0; off >>= 1) { if (tid < off) red[tid] += red[tid + off]; __syncthreads(); }
    float mean = red[0] * (1.f / Dl);
    __syncthreads();

    float v = 0.f;
    #pragma unroll
    for (int j = 0; j < 4; ++j) { float d = lx[j] - mean; v += d * d; }
    red[tid] = v; __syncthreads();
    for (int off = 128; off > 0; off >>= 1) { if (tid < off) red[tid] += red[tid + off]; __syncthreads(); }
    float rstd = rsqrtf(red[0] * (1.f / Dl) + 1e-5f);

    #pragma unroll
    for (int j = 0; j < 4; ++j) {
        int c = tid + 256 * j;
        out[(size_t)row * Dl + c] = f2bf((lx[j] - mean) * rstd * g[c] + be[c]);
    }
}

// ================= generic bf16 WMMA GEMM, 2x4 microtile =================
// C[M,N] = A[M,K] @ Bt[N,K]^T ; block = 128 thr (2x2 waves), block tile 64M x 128N
__global__ __launch_bounds__(128)
void k_gemm_bf16(const __bf16* __restrict__ A, int lda,
                 const __bf16* __restrict__ Bt, int ldbt,
                 float* Cf, __bf16* Cb, int ldc,
                 const float* __restrict__ bias,
                 const float* __restrict__ res, int ldres,
                 int K, int gelu) {
    constexpr int MI = 2, NI = 4;
    int lane = threadIdx.x & 31;
    int wav  = threadIdx.x >> 5;          // 0..3
    int wy = wav >> 1, wx = wav & 1;
    int mt = blockIdx.y * 64 + wy * 32;   // wave M base (2 tiles)
    int nt = blockIdx.x * 128 + wx * 64;  // wave N base (4 tiles)
    int mr   = lane & 15;
    int koff = (lane >> 4) * 8;

    v8f acc[MI][NI] = {};
    for (int k0 = 0; k0 < K; k0 += 32) {
        BFrag a[MI], b[NI];
        #pragma unroll
        for (int i = 0; i < MI; ++i) {
            const __bf16* ap = A + (size_t)(mt + i * 16 + mr) * lda + k0 + koff;
            a[i].q[0] = *(const uint4*)(ap);
            a[i].q[1] = *(const uint4*)(ap + 16);
        }
        #pragma unroll
        for (int j = 0; j < NI; ++j) {
            const __bf16* bp = Bt + (size_t)(nt + j * 16 + mr) * ldbt + k0 + koff;
            b[j].q[0] = *(const uint4*)(bp);
            b[j].q[1] = *(const uint4*)(bp + 16);
        }
        if (k0 + 32 < K) {
            __builtin_prefetch(A  + (size_t)(mt + mr) * lda  + k0 + 32 + koff, 0, 3);
            __builtin_prefetch(Bt + (size_t)(nt + mr) * ldbt + k0 + 32 + koff, 0, 3);
        }
        #pragma unroll
        for (int i = 0; i < MI; ++i)
            #pragma unroll
            for (int j = 0; j < NI; ++j)
                acc[i][j] = __builtin_amdgcn_wmma_f32_16x16x32_bf16(
                    false, a[i].v, false, b[j].v, (short)0, acc[i][j], false, false);
    }

    int rb = (lane >> 4) * 8;
    #pragma unroll
    for (int i = 0; i < MI; ++i) {
        #pragma unroll
        for (int j = 0; j < NI; ++j) {
            int col = nt + j * 16 + (lane & 15);
            #pragma unroll
            for (int r = 0; r < 8; ++r) {
                int row = mt + i * 16 + rb + r;
                float vv = acc[i][j][r];
                if (bias) vv += bias[col];
                if (gelu) vv = 0.5f * vv * (1.f + erff(vv * 0.70710678118f));
                if (res)  vv += res[(size_t)row * ldres + col];
                if (Cf) Cf[(size_t)row * ldc + col] = vv;
                else    Cb[(size_t)row * ldc + col] = f2bf(vv);
            }
        }
    }
}

// ================= fused attention: scores + softmax + AV =================
// One block (8 waves) owns a 16-row query strip for one (b,h):
//   phase 1: S[16][2048] = scale*Q K^T + prev   (S in LDS, f32)
//   phase 2: row softmax (exp kept unnormalized in LDS, inv folded into writes)
//   phase 3: O = S @ V via K-split across waves, ds_add_f32 reduction
__global__ __launch_bounds__(256)
void k_attn_fused(const __bf16* __restrict__ qkv, const __bf16* __restrict__ vT,
                  const float* __restrict__ prev, float* __restrict__ attn_out,
                  __bf16* __restrict__ aout) {
    extern __shared__ char smem[];
    float* S      = (float*)smem;                          // 16*2048
    float* O      = S + 16 * 2048;                         // 16*64
    float* red    = O + 16 * 64;                           // 256
    float* rowmax = red + 256;                             // 16
    float* rowinv = rowmax + 16;                           // 16

    int tid = threadIdx.x, lane = tid & 31, wav = tid >> 5;   // 8 waves
    int bh = blockIdx.y; int b = bh >> 4, h = bh & 15;
    int m0 = blockIdx.x * 16;
    int mr = lane & 15, koff = (lane >> 4) * 8;

    size_t qbase = (size_t)b * kT * kD3 + (size_t)h * kDH;
    size_t kbase = qbase + kD;

    // --- phase 1: scores ---
    BFrag qf[2];
    #pragma unroll
    for (int kc = 0; kc < 2; ++kc) {
        const __bf16* ap = qkv + qbase + (size_t)(m0 + mr) * kD3 + kc * 32 + koff;
        qf[kc].q[0] = *(const uint4*)(ap);
        qf[kc].q[1] = *(const uint4*)(ap + 16);
    }
    const float scale = 0.125f;                            // 1/sqrt(64)
    size_t pbase = (size_t)bh * kT * kT + (size_t)m0 * kT;
    int rb = (lane >> 4) * 8;

    for (int nc = 0; nc < 16; ++nc) {
        int nt = wav * 256 + nc * 16;
        v8f acc = {};
        #pragma unroll
        for (int kc = 0; kc < 2; ++kc) {
            BFrag bf;
            const __bf16* bp = qkv + kbase + (size_t)(nt + mr) * kD3 + kc * 32 + koff;
            bf.q[0] = *(const uint4*)(bp);
            bf.q[1] = *(const uint4*)(bp + 16);
            acc = __builtin_amdgcn_wmma_f32_16x16x32_bf16(
                false, qf[kc].v, false, bf.v, (short)0, acc, false, false);
        }
        int col = nt + (lane & 15);
        #pragma unroll
        for (int r = 0; r < 8; ++r) {
            int row = rb + r;
            S[row * 2048 + col] = acc[r] * scale + prev[pbase + (size_t)row * kT + col];
        }
    }
    __syncthreads();

    // --- phase 2: softmax (16 threads per row) ---
    int row = tid >> 4, cl = tid & 15;
    float m = -3.4e38f;
    for (int j = 0; j < 128; ++j) m = fmaxf(m, S[row * 2048 + cl + j * 16]);
    red[tid] = m; __syncthreads();
    if (cl == 0) {
        float mm = red[tid];
        #pragma unroll
        for (int j = 1; j < 16; ++j) mm = fmaxf(mm, red[tid + j]);
        rowmax[row] = mm;
    }
    __syncthreads();
    float rm = rowmax[row];
    float ssum = 0.f;
    for (int j = 0; j < 128; ++j) {
        float e = __expf(S[row * 2048 + cl + j * 16] - rm);
        S[row * 2048 + cl + j * 16] = e;
        ssum += e;
    }
    red[tid] = ssum; __syncthreads();
    if (cl == 0) {
        float ss = red[tid];
        #pragma unroll
        for (int j = 1; j < 16; ++j) ss += red[tid + j];
        rowinv[row] = 1.f / ss;
    }
    __syncthreads();

    // normalized attn to global (coalesced); zero O accumulators
    size_t obase = (size_t)bh * kT * kT + (size_t)m0 * kT;
    for (int idx = tid; idx < 16 * 2048; idx += 256)
        attn_out[obase + idx] = S[idx] * rowinv[idx >> 11];
    for (int idx = tid; idx < 16 * 64; idx += 256) O[idx] = 0.f;
    __syncthreads();

    // --- phase 3: O = S @ V, K split across 8 waves ---
    v8f oacc[4] = {};
    int kb0 = wav * 256;
    for (int kc = 0; kc < 8; ++kc) {
        int k0 = kb0 + kc * 32;
        BFrag af;
        const float* sp = S + mr * 2048 + k0 + koff;
        #pragma unroll
        for (int i = 0; i < 8; ++i) af.e[i]     = f2bf(sp[i]);
        #pragma unroll
        for (int i = 0; i < 8; ++i) af.e[8 + i] = f2bf(sp[16 + i]);
        #pragma unroll
        for (int j = 0; j < 4; ++j) {
            BFrag bf;
            const __bf16* bp = vT + ((size_t)bh * kDH + j * 16 + mr) * kT + k0 + koff;
            bf.q[0] = *(const uint4*)(bp);
            bf.q[1] = *(const uint4*)(bp + 16);
            oacc[j] = __builtin_amdgcn_wmma_f32_16x16x32_bf16(
                false, af.v, false, bf.v, (short)0, oacc[j], false, false);
        }
    }
    #pragma unroll
    for (int j = 0; j < 4; ++j) {
        int col = j * 16 + (lane & 15);
        #pragma unroll
        for (int r = 0; r < 8; ++r)
            atomicAdd(&O[(rb + r) * 64 + col], oacc[j][r]);
    }
    __syncthreads();

    // write attention output, merged heads: aout[(b*T + m0 + r)*D + h*64 + n]
    for (int idx = tid; idx < 16 * 64; idx += 256) {
        int r2 = idx >> 6, c2 = idx & 63;
        aout[((size_t)b * kT + m0 + r2) * kD + (size_t)h * kDH + c2] =
            f2bf(O[idx] * rowinv[r2]);
    }
}

// ================= TDM probe (not launched): async-tensor codegen =================
// This toolchain exposes the 6-arg (clang-23 / therock-10.0) form:
//   (u32x4 g0, i32x8 g1, i32x4 g2, i32x4 g3, i32x8 g4, i32 cpol)
__global__ void k_tdm_probe() {
    u32x4 g0 = {};
    i32x8 g1 = {};
    i32x4 g2 = {};
    i32x4 g3 = {};
    i32x8 g4 = {};
    __builtin_amdgcn_tensor_load_to_lds(g0, g1, g2, g3, g4, 0);
    __builtin_amdgcn_s_wait_tensorcnt(0);
}

// ================= launch =================
extern "C" void kernel_launch(void* const* d_in, const int* in_sizes, int n_in,
                              void* d_out, int out_size, void* d_ws, size_t ws_size,
                              hipStream_t stream) {
    const float* x      = (const float*)d_in[0];
    const float* prev   = (const float*)d_in[1];
    const float* W_qkv  = (const float*)d_in[2];
    const float* W_out  = (const float*)d_in[3];
    const float* W_ffn1 = (const float*)d_in[4];
    const float* b_ffn1 = (const float*)d_in[5];
    const float* W_ffn2 = (const float*)d_in[6];
    const float* b_ffn2 = (const float*)d_in[7];
    const float* g1  = (const float*)d_in[8];
    const float* be1 = (const float*)d_in[9];
    const float* g2  = (const float*)d_in[10];
    const float* be2 = (const float*)d_in[11];

    const int BT = kB * kT;                       // 4096 rows
    float* out_x    = (float*)d_out;              // [B,T,D]
    float* out_attn = out_x + (size_t)BT * kD;    // [B,H,T,T]

    // carve workspace
    char* wsp = (char*)d_ws;
    auto carve = [&](size_t bytes) { char* p = wsp; wsp += (bytes + 255) & ~(size_t)255; return p; };
    __bf16* wqkv_t = (__bf16*)carve((size_t)kD3 * kD   * 2);   // [3072,1024]
    __bf16* wout_t = (__bf16*)carve((size_t)kD  * kD   * 2);   // [1024,1024]
    __bf16* wff1_t = (__bf16*)carve((size_t)kDFF * kD  * 2);   // [4096,1024]
    __bf16* wff2_t = (__bf16*)carve((size_t)kD  * kDFF * 2);   // [1024,4096]
    __bf16* xn     = (__bf16*)carve((size_t)BT * kD    * 2);   // LN output (reused)
    __bf16* qkv_b  = (__bf16*)carve((size_t)BT * kD3   * 2);
    __bf16* vT     = (__bf16*)carve((size_t)kB * kH * kDH * kT * 2);
    __bf16* aout_b = (__bf16*)carve((size_t)BT * kD    * 2);
    float*  x1     = (float*) carve((size_t)BT * kD    * 4);
    __bf16* ff_b   = (__bf16*)carve((size_t)BT * kDFF  * 2);

    const int SMEM = (16 * 2048 + 16 * 64 + 256 + 16 + 16) * 4;   // 136320 B
    (void)hipFuncSetAttribute((const void*)k_attn_fused,
                              hipFuncAttributeMaxDynamicSharedMemorySize, SMEM);

    // 1) weight convert + transpose
    k_cvt_transpose<<<dim3(kD3  / 32, kD   / 32), 256, 0, stream>>>(W_qkv,  wqkv_t, kD,   kD3);
    k_cvt_transpose<<<dim3(kD   / 32, kD   / 32), 256, 0, stream>>>(W_out,  wout_t, kD,   kD);
    k_cvt_transpose<<<dim3(kDFF / 32, kD   / 32), 256, 0, stream>>>(W_ffn1, wff1_t, kD,   kDFF);
    k_cvt_transpose<<<dim3(kD   / 32, kDFF / 32), 256, 0, stream>>>(W_ffn2, wff2_t, kDFF, kD);

    // 2) LN1
    k_ln_bf16<<<BT, 256, 0, stream>>>(x, g1, be1, xn);

    // 3) QKV GEMM -> bf16 [B*T, 3D]
    k_gemm_bf16<<<dim3(kD3 / 128, BT / 64), 128, 0, stream>>>(
        xn, kD, wqkv_t, kD, nullptr, qkv_b, kD3, nullptr, nullptr, 0, kD, 0);

    // 4) V transpose -> vT[bh][n][k]
    k_transpose_v<<<dim3(kT / 64, 1, kB * kH), 256, 0, stream>>>(qkv_b, vT);

    // 5) fused attention (writes post-softmax attn + head-merged context)
    k_attn_fused<<<dim3(kT / 16, kB * kH), 256, SMEM, stream>>>(
        qkv_b, vT, prev, out_attn, aout_b);

    // 6) out-proj + residual x -> x1 (f32)
    k_gemm_bf16<<<dim3(kD / 128, BT / 64), 128, 0, stream>>>(
        aout_b, kD, wout_t, kD, x1, nullptr, kD, nullptr, x, kD, kD, 0);

    // 7) LN2
    k_ln_bf16<<<BT, 256, 0, stream>>>(x1, g2, be2, xn);

    // 8) FFN1 + bias + GELU -> ff_b (bf16)
    k_gemm_bf16<<<dim3(kDFF / 128, BT / 64), 128, 0, stream>>>(
        xn, kD, wff1_t, kD, nullptr, ff_b, kDFF, b_ffn1, nullptr, 0, kD, 1);

    // 9) FFN2 + bias + residual x1 -> out_x (f32)
    k_gemm_bf16<<<dim3(kD / 128, BT / 64), 128, 0, stream>>>(
        ff_b, kDFF, wff2_t, kDFF, out_x, nullptr, kD, b_ffn2, x1, kD, kDFF, 0);
}
